// WindowAttention_6322191859788
// MI455X (gfx1250) — matrix operand order repaired and checked by reading, not compiled, not run on hardware
//
#include <hip/hip_runtime.h>
#include <hip/hip_bf16.h>
#include <math.h>

typedef __attribute__((ext_vector_type(16))) _Float16 v16h;
typedef __attribute__((ext_vector_type(8)))  _Float16 v8h;
typedef __attribute__((ext_vector_type(8)))  float    v8f;
typedef __attribute__((__vector_size__(16))) int i32x4;

#define C 256
#define L 64
#define HEADS 8
#define NW 2048
#define EPS 1e-5f

// strides (in elements)
#define XS  260   // f32 stride for sX / sR
#define HS  264   // f16 stride for 256-wide f16 arrays
#define KTS 72    // f16 stride for sKt (64 tokens + pad)
#define PS  72    // f16 stride for attention prob rows

// LDS layout (bytes)
#define OFF_X    0          // f32 [64][260]  raw window           66560 B
#define OFF_XH   66560      // f16 [64][264]  normalized x         33792 B
#define OFF_Q    100352     // f16 [64][264]  Q (pre-scaled)       33792 B
#define OFF_KT   134144     // f16 [256][72]  K^T  (later sO)      36864 B
#define OFF_V    171008     // f16 [64][264]  V                    33792 B
#define OFF_P    204800     // f16 [8][64][72] probs (later sR)    73728 B
#define OFF_MEAN 278528     // f32 [64]
#define OFF_RSTD 278784     // f32 [64]
#define OFF_RED  279040     // float2 [256]
#define SMEM_BYTES 281088

// CDNA5 async global->LDS path (ASYNCcnt-tracked, no VGPR round trip).
#if __has_builtin(__builtin_amdgcn_global_load_async_to_lds_b128) && \
    __has_builtin(__builtin_amdgcn_s_wait_asynccnt)
#define HAVE_ASYNC_LDS 1
#else
#define HAVE_ASYNC_LDS 0
#endif

__device__ __forceinline__ v16h cat8(v8h lo, v8h hi) {
  return __builtin_shufflevector(lo, hi, 0,1,2,3,4,5,6,7,8,9,10,11,12,13,14,15);
}

// A fragment (16x32 f16): lane holds row m0+(lane&15); lanes 0-15 take K chunks
// {0..7,16..23}, lanes 16-31 take {8..15,24..31}.  Two 16B loads.
__device__ __forceinline__ v16h load_a16(const _Float16* base, int stride,
                                         int m0, int k0, int lane) {
  const _Float16* p = base + (m0 + (lane & 15)) * stride + k0 + ((lane >> 4) << 3);
  v8h lo = *(const v8h*)p;
  v8h hi = *(const v8h*)(p + 16);
  return cat8(lo, hi);
}

// B fragment (32x16 f16, K-major memory): lane holds col n0+(lane&15); lanes 0-15 take K
// {0..7,16..23}, lanes 16-31 take {8..15,24..31}.
__device__ __forceinline__ v16h load_b16(const _Float16* base, int stride,
                                         int k0, int n0, int lane) {
  const _Float16* p = base + (k0 + ((lane >> 4) << 3)) * stride + n0 + (lane & 15);
  v16h r;
  #pragma unroll
  for (int i = 0; i < 8; i++) {
    r[i]     = p[i * stride];
    r[8 + i] = p[(16 + i) * stride];
  }
  return r;
}

__device__ __forceinline__ v8f wmma16(v16h a, v16h b, v8f c) {
  return __builtin_amdgcn_wmma_f32_16x16x32_f16(false, a, false, b,
                                                (short)0, c, false, false);
}

// ---- prep: fold LN affine gains into QKV weights, transpose all weights to
// [k][n] f16 (K-major matches WMMA B-fragment striping: lane=K row).
__global__ __launch_bounds__(256)
void swin_prep_w(const float* __restrict__ w_in, const float* __restrict__ w_out,
                 const float* __restrict__ g_q, const float* __restrict__ g_k,
                 const float* __restrict__ g_v, _Float16* __restrict__ Wt) {
  int idx = blockIdx.x * 256 + threadIdx.x;   // 4*65536 total
  int mtx = idx >> 16;
  int k   = (idx >> 8) & 255;
  int n   = idx & 255;
  float w;
  if (mtx < 3) {
    const float* g = (mtx == 0) ? g_q : ((mtx == 1) ? g_k : g_v);
    w = w_in[(mtx * 256 + n) * 256 + k] * g[k];
  } else {
    w = w_out[n * 256 + k];
  }
  Wt[idx] = (_Float16)w;
}

// fold LN affine biases through the projection: b'[n] = b_in[n] + sum_k W[n,k]*b_ln[k]
__global__ __launch_bounds__(256)
void swin_prep_b(const float* __restrict__ w_in, const float* __restrict__ b_in,
                 const float* __restrict__ b_out, const float* __restrict__ b_q,
                 const float* __restrict__ b_k, const float* __restrict__ b_v,
                 float* __restrict__ biases) {
  int n = blockIdx.x * 256 + threadIdx.x;     // 1024 total
  if (n < 768) {
    int mtx = n >> 8;
    const float* bv = (mtx == 0) ? b_q : ((mtx == 1) ? b_k : b_v);
    float s = b_in[n];
    for (int k = 0; k < 256; k++) s += w_in[n * 256 + k] * bv[k];
    biases[n] = s;
  } else {
    biases[n] = b_out[n - 768];
  }
}

// ---- main: one workgroup (8 wave32) per 8x8 window.
__global__ __launch_bounds__(256)
void swin_win_attn(const float* __restrict__ xin,
                   const _Float16* __restrict__ Wt,   // [4][256k][256n] f16
                   const float* __restrict__ biases,  // [1024] folded biases
                   const float* __restrict__ g_o, const float* __restrict__ b_o,
                   float* __restrict__ yout) {
  extern __shared__ char smem[];
  float*    sX  = (float*)(smem + OFF_X);
  _Float16* sXH = (_Float16*)(smem + OFF_XH);
  _Float16* sQ  = (_Float16*)(smem + OFF_Q);
  _Float16* sKT = (_Float16*)(smem + OFF_KT);
  _Float16* sO  = (_Float16*)(smem + OFF_KT);   // alias: live after sKT dead
  _Float16* sV  = (_Float16*)(smem + OFF_V);
  _Float16* sP  = (_Float16*)(smem + OFF_P);
  float*    sR  = (float*)(smem + OFF_P);       // alias: live after sP dead
  float*    sMean = (float*)(smem + OFF_MEAN);
  float*    sRstd = (float*)(smem + OFF_RSTD);
  float2*   sRed  = (float2*)(smem + OFF_RED);

  const int tid  = threadIdx.x;
  const int lane = tid & 31;
  const int wv   = tid >> 5;

  const int w  = blockIdx.x;
  const int b  = w >> 6;
  const int wy = (w >> 3) & 7;
  const int wx = w & 7;

  // hint weight tiles into cache early (L2-resident across all 2048 blocks)
  __builtin_prefetch(Wt + (wv << 12), 0, 1);

  // ---- Phase 1: gather window -> LDS (64 tokens x 256 f32)
#if HAVE_ASYNC_LDS
  for (int i = tid; i < L * 64; i += 256) {
    int t  = i >> 6;
    int p4 = i & 63;
    int gy = (wy << 3) + (t >> 3);
    int gx = (wx << 3) + (t & 7);
    size_t ga = ((((size_t)b * 64 + gy) * 64 + gx) * C) + ((size_t)p4 << 2);
    __builtin_amdgcn_global_load_async_to_lds_b128(
        (__attribute__((address_space(1))) i32x4*)(xin + ga),
        (__attribute__((address_space(3))) i32x4*)(sX + t * XS + (p4 << 2)),
        0, 0);
  }
  __builtin_amdgcn_s_wait_asynccnt(0);
#else
  for (int i = tid; i < L * 64; i += 256) {
    int t  = i >> 6;
    int p4 = i & 63;
    int gy = (wy << 3) + (t >> 3);
    int gx = (wx << 3) + (t & 7);
    size_t ga = ((((size_t)b * 64 + gy) * 64 + gx) * C) + ((size_t)p4 << 2);
    float4 v = *(const float4*)(xin + ga);
    *(float4*)(sX + t * XS + (p4 << 2)) = v;
  }
#endif
  __syncthreads();

  // ---- Phase 1b: LN stats shared by q/k/v (affine folded into weights)
  {
    int t = tid >> 2, q = tid & 3;
    float s = 0.f, s2 = 0.f;
    const float* row = sX + t * XS + q * 64;
    #pragma unroll 16
    for (int c = 0; c < 64; c++) { float v = row[c]; s += v; s2 += v * v; }
    sRed[tid] = make_float2(s, s2);
  }
  __syncthreads();
  if (tid < L) {
    float s = 0.f, s2 = 0.f;
    #pragma unroll
    for (int i = 0; i < 4; i++) { float2 r = sRed[tid * 4 + i]; s += r.x; s2 += r.y; }
    float m = s * (1.f / 256.f);
    float v = s2 * (1.f / 256.f) - m * m;
    sMean[tid] = m;
    sRstd[tid] = rsqrtf(v + EPS);
  }
  __syncthreads();
  {
    int t = tid >> 2, q = tid & 3;
    float m = sMean[t], rs = sRstd[t];
    const float* row = sX + t * XS + q * 64;
    _Float16* orow = sXH + t * HS + q * 64;
    #pragma unroll 16
    for (int c = 0; c < 64; c++) orow[c] = (_Float16)((row[c] - m) * rs);
  }
  __syncthreads();

  // ---- Phase 2: packed QKV projection, 192 tiles of 16x16, K=256
  for (int t = wv; t < 192; t += 8) {
    int mt = t & 3;
    int nt = t >> 2;
    int m0 = mt << 4;
    int ng = nt << 4;            // 0..767 across Q|K|V
    int mtx = ng >> 8;
    int nl  = ng & 255;
    const _Float16* wmat = Wt + (mtx << 16);
    v8f acc = {};
    #pragma unroll
    for (int kk = 0; kk < 8; kk++) {
      v16h a   = load_a16(sXH, HS, m0, kk << 5, lane);
      v16h bfr = load_b16(wmat, 256, kk << 5, nl, lane);
      acc = wmma16(a, bfr, acc);
    }
    float bias = biases[ng + (lane & 15)];
    int col = nl + (lane & 15);
    int rb  = m0 + ((lane >> 4) << 3);
    if (mtx == 0) {              // Q, pre-scaled by 1/sqrt(d)
      #pragma unroll
      for (int j = 0; j < 8; j++)
        sQ[(rb + j) * HS + col] = (_Float16)((acc[j] + bias) * 0.1767766953f);
    } else if (mtx == 1) {       // K stored transposed: [channel][token]
      #pragma unroll
      for (int j = 0; j < 8; j++)
        sKT[col * KTS + rb + j] = (_Float16)(acc[j] + bias);
    } else {                     // V row-major
      #pragma unroll
      for (int j = 0; j < 8; j++)
        sV[(rb + j) * HS + col] = (_Float16)(acc[j] + bias);
    }
  }
  __syncthreads();

  // ---- Phase 3: per-head scores + softmax (wave wv == head wv)
  const int h = wv;
  _Float16* sPh = sP + h * (L * PS);
  {
    v16h aq[4], bk[4];
    #pragma unroll
    for (int mt = 0; mt < 4; mt++) aq[mt] = load_a16(sQ, HS, mt << 4, h << 5, lane);
    #pragma unroll
    for (int nt = 0; nt < 4; nt++) bk[nt] = load_b16(sKT, KTS, h << 5, nt << 4, lane);
    v8f sc[4][4];
    #pragma unroll
    for (int mt = 0; mt < 4; mt++)
      #pragma unroll
      for (int nt = 0; nt < 4; nt++) {
        v8f z = {};
        sc[mt][nt] = wmma16(aq[mt], bk[nt], z);   // K = d = 32: one WMMA/tile
      }
    // softmax: row m lives in 16 lanes (one half-wave) x 4 nt registers
    #pragma unroll
    for (int mt = 0; mt < 4; mt++) {
      #pragma unroll
      for (int j = 0; j < 8; j++) {
        float mx = -3.4e38f;
        #pragma unroll
        for (int nt = 0; nt < 4; nt++) mx = fmaxf(mx, sc[mt][nt][j]);
        #pragma unroll
        for (int d = 1; d < 16; d <<= 1) mx = fmaxf(mx, __shfl_xor(mx, d, 32));
        float e[4];
        float sum = 0.f;
        #pragma unroll
        for (int nt = 0; nt < 4; nt++) { e[nt] = __expf(sc[mt][nt][j] - mx); sum += e[nt]; }
        #pragma unroll
        for (int d = 1; d < 16; d <<= 1) sum += __shfl_xor(sum, d, 32);
        float inv = 1.f / sum;
        int m = (mt << 4) + j + ((lane >> 4) << 3);
        #pragma unroll
        for (int nt = 0; nt < 4; nt++)
          sPh[m * PS + (nt << 4) + (lane & 15)] = (_Float16)(e[nt] * inv);
      }
    }
  }
  __syncthreads();   // all sKT reads done -> safe to overwrite as sO

  // ---- Phase 4: O_head = P @ V   (4x2 tiles, K = 64 tokens)
  #pragma unroll
  for (int mt = 0; mt < 4; mt++) {
    #pragma unroll
    for (int nc = 0; nc < 2; nc++) {
      v8f acc = {};
      #pragma unroll
      for (int kk = 0; kk < 2; kk++) {
        v16h a   = load_a16(sPh, PS, mt << 4, kk << 5, lane);
        v16h bfr = load_b16(sV, HS, kk << 5, (h << 5) + (nc << 4), lane);
        acc = wmma16(a, bfr, acc);
      }
      int col = (h << 5) + (nc << 4) + (lane & 15);
      int rb  = (mt << 4) + ((lane >> 4) << 3);
      #pragma unroll
      for (int j = 0; j < 8; j++)
        sO[(rb + j) * HS + col] = (_Float16)acc[j];
    }
  }
  __syncthreads();   // all sP reads done -> safe to overwrite as sR

  // ---- Phase 5: out projection, 64 tiles, K=256, f32 result -> sR
  for (int t = wv; t < 64; t += 8) {
    int mt = t & 3;
    int nt = t >> 2;
    int m0 = mt << 4;
    int n0 = nt << 4;
    const _Float16* wmat = Wt + (3 << 16);
    v8f acc = {};
    #pragma unroll
    for (int kk = 0; kk < 8; kk++) {
      v16h a   = load_a16(sO, HS, m0, kk << 5, lane);
      v16h bfr = load_b16(wmat, 256, kk << 5, n0, lane);
      acc = wmma16(a, bfr, acc);
    }
    float bias = biases[768 + n0 + (lane & 15)];
    int col = n0 + (lane & 15);
    int rb  = m0 + ((lane >> 4) << 3);
    #pragma unroll
    for (int j = 0; j < 8; j++)
      sR[(rb + j) * XS + col] = acc[j] + bias;
  }
  __syncthreads();

  // ---- Phase 6: residual + final LayerNorm + window-merge store
  {
    int t = tid >> 2, q = tid & 3;
    float s = 0.f, s2 = 0.f;
    float* rrow = sR + t * XS + q * 64;
    const float* xrow = sX + t * XS + q * 64;
    #pragma unroll 16
    for (int c = 0; c < 64; c++) {
      float r = xrow[c] + rrow[c];
      rrow[c] = r;
      s += r; s2 += r * r;
    }
    sRed[tid] = make_float2(s, s2);
  }
  __syncthreads();
  if (tid < L) {
    float s = 0.f, s2 = 0.f;
    #pragma unroll
    for (int i = 0; i < 4; i++) { float2 r = sRed[tid * 4 + i]; s += r.x; s2 += r.y; }
    float m = s * (1.f / 256.f);
    float v = s2 * (1.f / 256.f) - m * m;
    sMean[tid] = m;
    sRstd[tid] = rsqrtf(v + EPS);
  }
  __syncthreads();
  {
    int t = tid >> 2, q = tid & 3;
    float m = sMean[t], rs = sRstd[t];
    int gy = (wy << 3) + (t >> 3);
    int gx = (wx << 3) + (t & 7);
    size_t gbase = (((size_t)b * 64 + gy) * 64 + gx) * C;
    const float* rrow = sR + t * XS;
    #pragma unroll 4
    for (int c4 = q * 16; c4 < q * 16 + 16; c4++) {
      int c = c4 << 2;
      float4 r  = *(const float4*)(rrow + c);
      float4 g  = *(const float4*)(g_o + c);
      float4 bb = *(const float4*)(b_o + c);
      float4 y;
      y.x = (r.x - m) * rs * g.x + bb.x;
      y.y = (r.y - m) * rs * g.y + bb.y;
      y.z = (r.z - m) * rs * g.z + bb.z;
      y.w = (r.w - m) * rs * g.w + bb.w;
      *(float4*)(yout + gbase + c) = y;
    }
  }
}

extern "C" void kernel_launch(void* const* d_in, const int* in_sizes, int n_in,
                              void* d_out, int out_size, void* d_ws, size_t ws_size,
                              hipStream_t stream) {
  (void)in_sizes; (void)n_in; (void)out_size; (void)ws_size;
  const float* x     = (const float*)d_in[0];
  const float* w_in  = (const float*)d_in[1];
  const float* b_in  = (const float*)d_in[2];
  const float* w_out = (const float*)d_in[3];
  const float* b_out = (const float*)d_in[4];
  const float* g_q   = (const float*)d_in[5];
  const float* b_q   = (const float*)d_in[6];
  const float* g_k   = (const float*)d_in[7];
  const float* b_k   = (const float*)d_in[8];
  const float* g_v   = (const float*)d_in[9];
  const float* b_v   = (const float*)d_in[10];
  const float* g_o   = (const float*)d_in[11];
  const float* b_o   = (const float*)d_in[12];
  float* y = (float*)d_out;

  _Float16* Wt   = (_Float16*)d_ws;                       // 4*65536 f16 = 512KB
  float*   biases = (float*)((char*)d_ws + 4 * 65536 * 2); // 1024 f32

  swin_prep_w<<<1024, 256, 0, stream>>>(w_in, w_out, g_q, g_k, g_v, Wt);
  swin_prep_b<<<4, 256, 0, stream>>>(w_in, b_in, b_out, b_q, b_k, b_v, biases);

  (void)hipFuncSetAttribute((const void*)swin_win_attn,
                            hipFuncAttributeMaxDynamicSharedMemorySize,
                            SMEM_BYTES);
  swin_win_attn<<<NW, 256, SMEM_BYTES, stream>>>(x, Wt, biases, g_o, b_o, y);
}
